// LearnerForAttention_6081673691349
// MI455X (gfx1250) — matrix operand order repaired and checked by reading, not compiled, not run on hardware
//
#include <hip/hip_runtime.h>
#include <hip/hip_bf16.h>

typedef __attribute__((ext_vector_type(16))) _Float16 v16h;
typedef __attribute__((ext_vector_type(8)))  _Float16 v8h;
typedef __attribute__((ext_vector_type(8)))  float    v8f;
typedef __attribute__((ext_vector_type(2)))  float    v2f;

#define LSTM_H   20
#define KCONST   7.0f
#define EPS_LS   1e-7f
#define EXP_K    1096.6331584284585f   // e^7

__device__ __forceinline__ float bcast_lane(float v, int lane) {
    return __int_as_float(__builtin_amdgcn_readlane(__float_as_int(v), lane));
}

#if __has_builtin(__builtin_amdgcn_tanhf)
__device__ __forceinline__ float fast_tanh(float x) {
    return __builtin_amdgcn_tanhf(x);          // native v_tanh_f32 (gfx1250 TRANS)
}
#else
__device__ __forceinline__ float fast_tanh(float x) {
    return 1.0f - 2.0f * __builtin_amdgcn_rcpf(__expf(2.0f * x) + 1.0f);
}
#endif

__device__ __forceinline__ float fast_sigmoid(float x) {
    // sigmoid(x) = 0.5*tanh(x/2) + 0.5  -> mul + tanh + fma
    return fmaf(0.5f, fast_tanh(0.5f * x), 0.5f);
}

// Single-wave sequential LSTM scan over N steps, hidden=20.
// - gate matvec uses v_pk_fma_f32 (gate-pair packing, h broadcast via v_readlane)
// - output FC amortized via v_wmma_f32_16x16x32_f16 every 16 steps
__global__ __launch_bounds__(32)
void lstm_scan_kernel(const float* __restrict__ inp,
                      const float* __restrict__ cWih, const float* __restrict__ cWhh,
                      const float* __restrict__ cbih, const float* __restrict__ cbhh,
                      const float* __restrict__ cfw,  const float* __restrict__ cfb,
                      const float* __restrict__ fWih, const float* __restrict__ fWhh,
                      const float* __restrict__ fbih, const float* __restrict__ fbhh,
                      const float* __restrict__ ffw,  const float* __restrict__ ffb,
                      const int*   __restrict__ is_conv,
                      float* __restrict__ out, int N)
{
    const int lane = threadIdx.x;           // 0..31, wave32
    const bool conv = (is_conv[0] != 0);
    const float* Wih = conv ? cWih : fWih;  // (80,2) row-major
    const float* Whh = conv ? cWhh : fWhh;  // (80,20) row-major
    const float* bih = conv ? cbih : fbih;  // (80,)
    const float* bhh = conv ? cbhh : fbhh;  // (80,)
    const float* fw  = conv ? cfw  : ffw;   // (1,20)
    const float* fb  = conv ? cfb  : ffb;   // (1,)

    // Lane j (<20) owns hidden unit j. Gate rows: i=j, f=20+j, g=40+j, o=60+j.
    const int j = (lane < LSTM_H) ? lane : 0;
    const int r0 = j, r1 = 20 + j, r2 = 40 + j, r3 = 60 + j;

    // Packed weights: pair01 = (i,f) gates, pair23 = (g,o) gates.
    v2f w01[LSTM_H], w23[LSTM_H];
#pragma unroll
    for (int k = 0; k < LSTM_H; ++k) {
        w01[k] = (v2f){ Whh[r0 * LSTM_H + k], Whh[r1 * LSTM_H + k] };
        w23[k] = (v2f){ Whh[r2 * LSTM_H + k], Whh[r3 * LSTM_H + k] };
    }
    const v2f wi0_01 = (v2f){ Wih[2 * r0], Wih[2 * r1] };
    const v2f wi0_23 = (v2f){ Wih[2 * r2], Wih[2 * r3] };
    const v2f wi1_01 = (v2f){ Wih[2 * r0 + 1], Wih[2 * r1 + 1] };
    const v2f wi1_23 = (v2f){ Wih[2 * r2 + 1], Wih[2 * r3 + 1] };
    const v2f bias01 = (v2f){ bih[r0] + bhh[r0], bih[r1] + bhh[r1] };
    const v2f bias23 = (v2f){ bih[r2] + bhh[r2], bih[r3] + bhh[r3] };
    const float fcb = fb[0];

    // WMMA B operand: B[k][n] = fcw_pad[k] for ALL columns n -> every column of
    // D equals the 16 outputs. Lanes 0-15 carry K={0..7,16..23}, lanes 16-31 K={8..15,24..31}.
    v16h bmat;
#pragma unroll
    for (int i = 0; i < 16; ++i) {
        const int kk = ((i < 8) ? i : (8 + i)) + ((lane < 16) ? 0 : 8);
        bmat[i] = (_Float16)((kk < LSTM_H) ? fw[kk] : 0.0f);
    }

    // A-tile staging: rows 0-15 = real 16x32 f16 tile (K 20..31 stay zero);
    // rows 16-31 = garbage dump for lanes 20-31 so the store is unconditional.
    __shared__ __align__(16) _Float16 atile[32][32];
    for (int idx = lane; idx < 32 * 32; idx += 32)
        (&atile[0][0])[idx] = (_Float16)0.0f;
    __syncthreads();
    _Float16* const stslot = &atile[(lane < LSTM_H) ? 0 : 16][lane];

    const float inv_k = 1.0f / KCONST;
    float hj = 0.0f, cj = 0.0f;

    for (int t0 = 0; t0 < N; t0 += 16) {
        if (t0 + 1024 < N)
            __builtin_prefetch(inp + t0 + 1024, 0, 0);   // global_prefetch_b8

#pragma unroll
        for (int tt = 0; tt < 16; ++tt) {
            // --- feature path (independent of recurrence; pipelines ahead) ---
            const float v  = inp[t0 + tt];
            float lg = __logf(__builtin_fabsf(v) + EPS_LS) * inv_k;
            lg = fmaxf(lg, -1.0f);
            const float sg = fminf(fmaxf(lg * EXP_K, -1.0f), 1.0f);

            // --- gate pre-activations: xg + W_hh @ h, packed (i,f)/(g,o) ---
            const v2f lg2 = (v2f){ lg, lg };
            const v2f sg2 = (v2f){ sg, sg };
            v2f ga01 = __builtin_elementwise_fma(lg2, wi0_01, bias01);
            v2f ga23 = __builtin_elementwise_fma(lg2, wi0_23, bias23);
            v2f gb01 = sg2 * wi1_01;
            v2f gb23 = sg2 * wi1_23;
#pragma unroll
            for (int k = 0; k < LSTM_H; k += 2) {
                const float hk0 = bcast_lane(hj, k);
                const float hk1 = bcast_lane(hj, k + 1);
                const v2f hv0 = (v2f){ hk0, hk0 };
                const v2f hv1 = (v2f){ hk1, hk1 };
                ga01 = __builtin_elementwise_fma(hv0, w01[k],     ga01);
                gb01 = __builtin_elementwise_fma(hv1, w01[k + 1], gb01);
                ga23 = __builtin_elementwise_fma(hv0, w23[k],     ga23);
                gb23 = __builtin_elementwise_fma(hv1, w23[k + 1], gb23);
            }
            const v2f g01 = ga01 + gb01;   // (i, f)
            const v2f g23 = ga23 + gb23;   // (g, o)

            const float i_ = fast_sigmoid(g01.x);
            const float f_ = fast_sigmoid(g01.y);
            const float g_ = fast_tanh   (g23.x);
            const float o_ = fast_sigmoid(g23.y);
            cj = fmaf(f_, cj, i_ * g_);
            hj = o_ * fast_tanh(cj);

            stslot[tt * 32] = (_Float16)hj;   // unconditional ds_store_b16
        }
        __syncthreads();   // order staging stores before cross-lane A reads

        // A operand per 16-bit 16x32 layout: lanes 0-15 -> K {0..7,16..23},
        // lanes 16-31 -> K {8..15,24..31}; two aligned b128 LDS reads.
        const int m     = lane & 15;
        const int kbase = (lane < 16) ? 0 : 8;
        const v8h lo = *reinterpret_cast<const v8h*>(&atile[m][kbase]);
        const v8h hi = *reinterpret_cast<const v8h*>(&atile[m][16 + kbase]);
        const v16h amat = __builtin_shufflevector(lo, hi,
            0, 1, 2, 3, 4, 5, 6, 7, 8, 9, 10, 11, 12, 13, 14, 15);

        v8f acc = {};
        acc = __builtin_amdgcn_wmma_f32_16x16x32_f16(
            false, amat, false, bmat, (short)0, acc, false, false);

        // D: lanes 0-15 hold M=0..7 in acc[0..7]; lanes 16-31 hold M=8..15.
        // Bias add in all lanes (uniform work, no duplication in branch arms),
        // then a single guarded 32-byte store from lanes 0 and 16.
        v8f res;
#pragma unroll
        for (int r = 0; r < 8; ++r) res[r] = acc[r] + fcb;
        if ((lane & 15) == 0) {
            float* p = out + t0 + ((lane >> 1) & 8);   // lane 0 -> +0, lane 16 -> +8
#pragma unroll
            for (int r = 0; r < 8; ++r) p[r] = res[r];
        }
        // no trailing barrier needed: single wave, DS ops are in-order and the
        // next tile's stores may-alias these loads, so ordering is preserved.
    }
}

extern "C" void kernel_launch(void* const* d_in, const int* in_sizes, int n_in,
                              void* d_out, int out_size, void* d_ws, size_t ws_size,
                              hipStream_t stream) {
    (void)n_in; (void)d_ws; (void)ws_size; (void)out_size;
    const float* inp  = (const float*)d_in[0];
    const float* cWih = (const float*)d_in[1];
    const float* cWhh = (const float*)d_in[2];
    const float* cbih = (const float*)d_in[3];
    const float* cbhh = (const float*)d_in[4];
    const float* cfw  = (const float*)d_in[5];
    const float* cfb  = (const float*)d_in[6];
    const float* fWih = (const float*)d_in[7];
    const float* fWhh = (const float*)d_in[8];
    const float* fbih = (const float*)d_in[9];
    const float* fbhh = (const float*)d_in[10];
    const float* ffw  = (const float*)d_in[11];
    const float* ffb  = (const float*)d_in[12];
    const int*   iscv = (const int*)d_in[13];

    const int N = in_sizes[0];   // 1024*1024, divisible by 16
    lstm_scan_kernel<<<1, 32, 0, stream>>>(inp, cWih, cWhh, cbih, cbhh, cfw, cfb,
                                           fWih, fWhh, fbih, fbhh, ffw, ffb,
                                           iscv, (float*)d_out, N);
}
